// CrossCorrMambaReasoner_39264591020659
// MI455X (gfx1250) — matrix operand order repaired and checked by reading, not compiled, not run on hardware
//
#include <hip/hip_runtime.h>
#include <hip/hip_bf16.h>

typedef _Float16 f16;
typedef __attribute__((ext_vector_type(16))) _Float16 v16h;
typedef __attribute__((ext_vector_type(8)))  _Float16 v8h;
typedef __attribute__((ext_vector_type(8)))  float    v8f;

#define QN   75
#define SN   5
#define PN   25
#define KK2  25
#define CN   640
#define RN   9375    /* 75*5*25 rows */
#define HIDN 256
#define DIN  512
#define DSTN 16
#define SEQL 25
#define BNCH 375

// ---------------------------------------------------------------- helpers
__device__ __forceinline__ v8h ld8(const f16* p) { return *(const v8h*)p; }

#define SHUF16(lo, hi) __builtin_shufflevector((lo), (hi), 0,1,2,3,4,5,6,7,8,9,10,11,12,13,14,15)

// ---------------------------------------------------------------- weight transpose+pack: W[K][N] f32 -> Wt[N][Kpad] f16 (zero padded)
__global__ void k_wt(const float* __restrict__ W, f16* __restrict__ Wt, int K, int N, int Kpad) {
  int t = blockIdx.x * blockDim.x + threadIdx.x;
  if (t >= N * Kpad) return;
  int n = t / Kpad, k = t % Kpad;
  float v = (k < K) ? W[(size_t)k * N + n] : 0.0f;
  Wt[t] = (f16)v;
}

// ---------------------------------------------------------------- channel-vector norms for query and support
__global__ void k_norms(const float* __restrict__ query, const float* __restrict__ support,
                        float* __restrict__ qnorm, float* __restrict__ snorm) {
  __shared__ float red[64];
  int idx = blockIdx.x, tid = threadIdx.x;
  const float* base;
  if (idx < QN * PN) base = query + (size_t)(idx / PN) * (CN * PN) + (idx % PN);
  else { int k = idx - QN * PN; base = support + (size_t)(k / PN) * (CN * PN) + (k % PN); }
  float ss = 0.f;
  for (int c = tid; c < CN; c += 64) { float v = base[(size_t)c * PN]; ss += v * v; }
  red[tid] = ss; __syncthreads();
  for (int s = 32; s > 0; s >>= 1) { if (tid < s) red[tid] += red[tid + s]; __syncthreads(); }
  if (tid == 0) {
    float n = fmaxf(sqrtf(red[0]), 1e-8f);
    if (idx < QN * PN) qnorm[idx] = n; else snorm[idx - QN * PN] = n;
  }
}

// qn16[p][q][c] = query[q][c][p] / qnorm
__global__ void k_pack_q(const float* __restrict__ query, const float* __restrict__ qnorm,
                         f16* __restrict__ qn16) {
  int t = blockIdx.x * blockDim.x + threadIdx.x;
  if (t >= PN * QN * CN) return;
  int c = t % CN, q = (t / CN) % QN, p = t / (CN * QN);
  qn16[t] = (f16)(query[(size_t)q * CN * PN + (size_t)c * PN + p] / qnorm[q * PN + p]);
}

// su16[s][p][k][c] = padded shifted patch of normalized support
__global__ void k_pack_su(const float* __restrict__ support, const float* __restrict__ snorm,
                          f16* __restrict__ su16) {
  int t = blockIdx.x * blockDim.x + threadIdx.x;
  if (t >= SN * PN * KK2 * CN) return;
  int c = t % CN, k = (t / CN) % KK2, p = (t / (CN * KK2)) % PN, s = t / (CN * KK2 * PN);
  int i = k / 5, j = k % 5, h = p / 5, w = p % 5;
  int hs = h + i - 2, ws2 = w + j - 2;
  float v = 0.f;
  if (hs >= 0 && hs < 5 && ws2 >= 0 && ws2 < 5) {
    int pos = hs * 5 + ws2;
    v = support[(size_t)s * CN * PN + (size_t)c * PN + pos] / snorm[s * PN + pos];
  }
  su16[t] = (f16)v;
}

// ---------------------------------------------------------------- batched cross-correlation GEMM (WMMA, 32x16 macro-tile)
// per (s,p): cc[q][k] = sum_c qn16[p][q][c] * su16[s][p][k][c]   M=75 N=25 K=640
__global__ void k_ccorr(const f16* __restrict__ qn16, const f16* __restrict__ su16,
                        f16* __restrict__ cc16) {
  int wid = threadIdx.x >> 5, lane = threadIdx.x & 31;
  int tile = blockIdx.x * (blockDim.x >> 5) + wid;
  if (tile >= 6) return;                // 3 m-tiles(32) * 2 n-tiles(16)
  int batch = blockIdx.y;               // s*25+p
  int s = batch / PN, p = batch % PN;
  int tm = tile >> 1, tn = tile & 1;
  int mbase = tm * 32, nbase = tn * 16;
  int l15 = lane & 15;
  int rowA0 = min(mbase + l15, QN - 1);        // clamp: garbage rows never stored
  int rowA1 = min(mbase + 16 + l15, QN - 1);
  int rowB  = min(nbase + l15, KK2 - 1);
  int koA = (lane < 16) ? 0 : 8;
  int koB = (lane < 16) ? 0 : 8;
  const f16* A0 = qn16 + (size_t)p * QN * CN + (size_t)rowA0 * CN + koA;
  const f16* A1 = qn16 + (size_t)p * QN * CN + (size_t)rowA1 * CN + koA;
  const f16* Bt = su16 + ((size_t)batch * KK2 + rowB) * CN + koB;
  v8f acc0 = {}, acc1 = {};
  for (int kb = 0; kb < CN; kb += 32) {
    v16h b  = SHUF16(ld8(Bt + kb), ld8(Bt + kb + 16));
    v16h a0 = SHUF16(ld8(A0 + kb), ld8(A0 + kb + 16));
    v16h a1 = SHUF16(ld8(A1 + kb), ld8(A1 + kb + 16));
    acc0 = __builtin_amdgcn_wmma_f32_16x16x32_f16(false, a0, false, b, (short)0, acc0, false, false);
    acc1 = __builtin_amdgcn_wmma_f32_16x16x32_f16(false, a1, false, b, (short)0, acc1, false, false);
  }
  int col = nbase + l15;
  int rb = mbase + ((lane < 16) ? 0 : 8);
#pragma unroll
  for (int r = 0; r < 8; ++r) {
    int q0 = rb + r, q1 = rb + 16 + r;
    float v0 = (col < KK2) ? acc0[r] : 0.f;   // zero the K-pad columns
    float v1 = (col < KK2) ? acc1[r] : 0.f;
    if (q0 < QN) cc16[(((size_t)q0 * SN + s) * PN + p) * 32 + col] = (f16)v0;
    if (q1 < QN) cc16[(((size_t)q1 * SN + s) * PN + p) * 32 + col] = (f16)v1;
  }
}

// ---------------------------------------------------------------- generic WMMA GEMM: C = A[MxK] * Bt[NxK]^T  (32x16 macro-tile/wave)
// flags: 1=+bias  2=relu  4=+resid  8=store f16  16=store f32
__global__ void k_gemm(const f16* __restrict__ A, int lda,
                       const f16* __restrict__ Bt, int ldb,
                       float* __restrict__ C, int ldc,
                       f16* __restrict__ C16, int ldc16,
                       int M, int N, int K,
                       const float* __restrict__ bias,
                       const float* __restrict__ resid, int ldres,
                       int flags) {
  int wid = threadIdx.x >> 5, lane = threadIdx.x & 31;
  int mt = (M + 31) >> 5, nt = (N + 15) >> 4;
  int tile = blockIdx.x * (blockDim.x >> 5) + wid;
  if (tile >= mt * nt) return;
  int tm = tile / nt, tn = tile % nt;
  int mbase = tm << 5, nbase = tn << 4;
  int l15 = lane & 15;
  int rowA0 = min(mbase + l15, M - 1);         // clamp: garbage rows never stored
  int rowA1 = min(mbase + 16 + l15, M - 1);
  int rowB  = min(nbase + l15, N - 1);
  int koA = (lane < 16) ? 0 : 8;
  int koB = (lane < 16) ? 0 : 8;
  const f16* Ab0 = A + (size_t)rowA0 * lda + koA;
  const f16* Ab1 = A + (size_t)rowA1 * lda + koA;
  const f16* Bb  = Bt + (size_t)rowB * ldb + koB;
  v8f acc0 = {}, acc1 = {};
  for (int kb = 0; kb < K; kb += 32) {
    v16h b  = SHUF16(ld8(Bb + kb), ld8(Bb + kb + 16));
    v16h a0 = SHUF16(ld8(Ab0 + kb), ld8(Ab0 + kb + 16));
    v16h a1 = SHUF16(ld8(Ab1 + kb), ld8(Ab1 + kb + 16));
    acc0 = __builtin_amdgcn_wmma_f32_16x16x32_f16(false, a0, false, b, (short)0, acc0, false, false);
    acc1 = __builtin_amdgcn_wmma_f32_16x16x32_f16(false, a1, false, b, (short)0, acc1, false, false);
  }
  int col = nbase + l15;
  int rb = mbase + ((lane < 16) ? 0 : 8);
  if (col < N) {
    float bv2 = (flags & 1) ? bias[col] : 0.f;
#pragma unroll
    for (int r = 0; r < 16; ++r) {
      int row = rb + ((r < 8) ? r : (8 + r));   // r 0..7 -> rb+r ; r 8..15 -> rb+16+(r-8)
      float av = (r < 8) ? acc0[r & 7] : acc1[r & 7];
      if (row < M) {
        float v = av + bv2;
        if (flags & 4) v += resid[(size_t)row * ldres + col];
        if (flags & 2) v = fmaxf(v, 0.f);
        if (flags & 16) C[(size_t)row * ldc + col] = v;
        if (flags & 8) C16[(size_t)row * ldc16 + col] = (f16)v;
      }
    }
  }
}

// ---------------------------------------------------------------- LayerNorm (cols=256) +optional relu +optional pos[row%25]
__global__ void k_ln(const float* __restrict__ X, const float* __restrict__ g,
                     const float* __restrict__ b, const float* __restrict__ pos,
                     float* __restrict__ Yf, f16* __restrict__ Y16, int relu) {
  __shared__ float red[256];
  int row = blockIdx.x, c = threadIdx.x;
  float x = X[(size_t)row * 256 + c];
  red[c] = x; __syncthreads();
  for (int s = 128; s > 0; s >>= 1) { if (c < s) red[c] += red[c + s]; __syncthreads(); }
  float mean = red[0] * (1.0f / 256.f); __syncthreads();
  float d0 = x - mean;
  red[c] = d0 * d0; __syncthreads();
  for (int s = 128; s > 0; s >>= 1) { if (c < s) red[c] += red[c + s]; __syncthreads(); }
  float var = red[0] * (1.0f / 256.f);
  float v = d0 * rsqrtf(var + 1e-5f) * g[c] + b[c];
  if (relu) v = fmaxf(v, 0.f);
  if (pos) v += pos[(row % SEQL) * 256 + c];
  if (Yf)  Yf[(size_t)row * 256 + c] = v;
  if (Y16) Y16[(size_t)row * 256 + c] = (f16)v;
}

// ---------------------------------------------------------------- causal depthwise conv (DCONV=4) + SiLU; reads xc half of xz
__global__ void k_conv(const float* __restrict__ xz, const float* __restrict__ conv_w,
                       const float* __restrict__ conv_b, float* __restrict__ xcf,
                       f16* __restrict__ xc16) {
  int t = blockIdx.x * blockDim.x + threadIdx.x;
  if (t >= RN * DIN) return;
  int r = t / DIN, d = t % DIN;
  int b = r / SEQL, l = r % SEQL;
  size_t base = (size_t)b * SEQL;
  float acc = conv_b[d];
#pragma unroll
  for (int k = 0; k < 4; ++k) {
    int ll = l + k - 3;
    if (ll >= 0) acc += xz[(base + ll) * 1024 + d] * conv_w[(size_t)d * 4 + k];
  }
  acc = acc / (1.f + __expf(-acc));   // silu: x*sigmoid(x)
  xcf[t] = acc;
  xc16[t] = (f16)acc;
}

// ---------------------------------------------------------------- pack dt GEMM input: xdbl[:, :16] -> [RN][32] f16 (zero padded)
__global__ void k_packdt(const float* __restrict__ xdbl, f16* __restrict__ dtin16) {
  int t = blockIdx.x * blockDim.x + threadIdx.x;
  if (t >= RN * 32) return;
  int k = t % 32;
  dtin16[t] = (k < 16) ? (f16)xdbl[(size_t)(t / 32) * 48 + k] : (f16)0.0f;
}

// ---------------------------------------------------------------- selective-scan: one block per sequence, one thread per channel d
__global__ void k_scan(const float* __restrict__ xdbl, const float* __restrict__ dtraw,
                       const float* __restrict__ xcf, const float* __restrict__ xz,
                       const float* __restrict__ A_log, const float* __restrict__ Dp,
                       f16* __restrict__ y16) {
  __shared__ float sB[DSTN], sC[DSTN];
  int b = blockIdx.x, d = threadIdx.x;
  float a[DSTN], h[DSTN];
#pragma unroll
  for (int n = 0; n < DSTN; ++n) { a[n] = -__expf(A_log[(size_t)d * DSTN + n]); h[n] = 0.f; }
  float Dd = Dp[d];
  for (int l = 0; l < SEQL; ++l) {
    size_t r = (size_t)b * SEQL + l;
    if (d < DSTN) sB[d] = xdbl[r * 48 + 16 + d];
    else if (d < 2 * DSTN) sC[d - DSTN] = xdbl[r * 48 + 32 + (d - DSTN)];
    __syncthreads();
    float dtr = dtraw[r * DIN + d];
    float dt = (dtr > 20.f) ? dtr : log1pf(__expf(dtr));   // softplus
    float xc = xcf[r * DIN + d];
    float y = 0.f;
#pragma unroll
    for (int n = 0; n < DSTN; ++n) {
      float dA = __expf(dt * a[n]);
      h[n] = dA * h[n] + dt * sB[n] * xc;
      y += h[n] * sC[n];
    }
    float z = xz[r * 1024 + DIN + d];
    float out = (y + Dd * xc) * (z / (1.f + __expf(-z)));
    y16[r * DIN + d] = (f16)out;
    __syncthreads();
  }
}

// ---------------------------------------------------------------- reverse sequence copy (within each group of 25 rows)
__global__ void k_rev(const float* __restrict__ in, float* __restrict__ out) {
  int t = blockIdx.x * blockDim.x + threadIdx.x;
  if (t >= RN * HIDN) return;
  int r = t / HIDN, c = t % HIDN;
  int g = r / SEQL, l = r % SEQL;
  out[(size_t)(g * SEQL + (SEQL - 1 - l)) * HIDN + c] = in[t];
}

// ---------------------------------------------------------------- concat [xf, reverse(xb)] -> f16 [RN][512]
__global__ void k_cat(const float* __restrict__ xf, const float* __restrict__ xbr,
                      f16* __restrict__ cat16) {
  int t = blockIdx.x * blockDim.x + threadIdx.x;
  if (t >= RN * 512) return;
  int r = t / 512, c = t % 512;
  float v;
  if (c < 256) v = xf[(size_t)r * 256 + c];
  else {
    int g = r / SEQL, l = r % SEQL;
    v = xbr[(size_t)(g * SEQL + (SEQL - 1 - l)) * 256 + (c - 256)];
  }
  cat16[t] = (f16)v;
}

// ---------------------------------------------------------------- attention score + softmax + pool + final LN (+labels)
__global__ void k_pool(const float* __restrict__ a1, const float* __restrict__ xfus,
                       const float* __restrict__ W2, const float* __restrict__ b2,
                       const float* __restrict__ fg, const float* __restrict__ fb,
                       float* __restrict__ out) {
  __shared__ float red[256];
  __shared__ float attn[SEQL];
  int g = blockIdx.x, tid = threadIdx.x;
  for (int l = 0; l < SEQL; ++l) {
    size_t r = (size_t)g * SEQL + l;
    red[tid] = a1[r * 256 + tid] * W2[tid];
    __syncthreads();
    for (int s = 128; s > 0; s >>= 1) { if (tid < s) red[tid] += red[tid + s]; __syncthreads(); }
    if (tid == 0) attn[l] = red[0] + b2[0];
    __syncthreads();
  }
  if (tid == 0) {
    float mx = attn[0];
    for (int l = 1; l < SEQL; ++l) mx = fmaxf(mx, attn[l]);
    float ssum = 0.f;
    for (int l = 0; l < SEQL; ++l) { float e = __expf(attn[l] - mx); attn[l] = e; ssum += e; }
    float inv = 1.f / ssum;
    for (int l = 0; l < SEQL; ++l) attn[l] *= inv;
  }
  __syncthreads();
  float pooled = 0.f;
  for (int l = 0; l < SEQL; ++l) pooled += xfus[((size_t)g * SEQL + l) * 256 + tid] * attn[l];
  // final LayerNorm over 256
  red[tid] = pooled; __syncthreads();
  for (int s = 128; s > 0; s >>= 1) { if (tid < s) red[tid] += red[tid + s]; __syncthreads(); }
  float mean = red[0] * (1.f / 256.f); __syncthreads();
  float d0 = pooled - mean;
  red[tid] = d0 * d0; __syncthreads();
  for (int s = 128; s > 0; s >>= 1) { if (tid < s) red[tid] += red[tid + s]; __syncthreads(); }
  float var = red[0] * (1.f / 256.f);
  out[(size_t)g * 256 + tid] = d0 * rsqrtf(var + 1e-5f) * fg[tid] + fb[tid];
  if (g == 0 && tid < 5) out[(size_t)QN * SN * HIDN + tid] = (float)tid;  // labels 0..4
}

// ================================================================ host
extern "C" void kernel_launch(void* const* d_in, const int* in_sizes, int n_in,
                              void* d_out, int out_size, void* d_ws, size_t ws_size,
                              hipStream_t stream) {
  (void)in_sizes; (void)n_in; (void)out_size; (void)ws_size;
  const float* support = (const float*)d_in[0];
  const float* query   = (const float*)d_in[1];
  const float* proj_W  = (const float*)d_in[2];
  const float* proj_b  = (const float*)d_in[3];
  const float* proj_g  = (const float*)d_in[4];
  const float* proj_bb = (const float*)d_in[5];
  const float* spos    = (const float*)d_in[6];
  const float* grp[2][13];
  for (int g = 0; g < 2; ++g)
    for (int i = 0; i < 13; ++i) grp[g][i] = (const float*)d_in[7 + g * 13 + i];
  const float* fus_W  = (const float*)d_in[33];
  const float* fus_b  = (const float*)d_in[34];
  const float* agg_W1 = (const float*)d_in[35];
  const float* agg_b1 = (const float*)d_in[36];
  const float* agg_W2 = (const float*)d_in[37];
  const float* agg_b2 = (const float*)d_in[38];
  const float* fin_g  = (const float*)d_in[39];
  const float* fin_b  = (const float*)d_in[40];
  float* out = (float*)d_out;

  char* wsb = (char*)d_ws;
  size_t off = 0;
  auto alloc = [&](size_t bytes) -> char* {
    char* p = wsb + off;
    off = (off + bytes + 255) & ~(size_t)255;
    return p;
  };
  float* qnorm  = (float*)alloc(QN * PN * 4);
  float* snorm  = (float*)alloc(SN * PN * 4);
  f16* qn16     = (f16*)alloc((size_t)PN * QN * CN * 2);
  f16* su16     = (f16*)alloc((size_t)SN * PN * KK2 * CN * 2);
  f16* cc16     = (f16*)alloc((size_t)RN * 32 * 2);
  float* x0     = (float*)alloc((size_t)RN * 256 * 4);
  float* tA     = (float*)alloc((size_t)RN * 256 * 4);
  float* tB     = (float*)alloc((size_t)RN * 256 * 4);
  float* xfF    = (float*)alloc((size_t)RN * 256 * 4);
  f16* h16      = (f16*)alloc((size_t)RN * 256 * 2);
  float* xz     = (float*)alloc((size_t)RN * 1024 * 4);
  float* xcf    = (float*)alloc((size_t)RN * 512 * 4);
  f16* xc16     = (f16*)alloc((size_t)RN * 512 * 2);
  float* xdbl   = (float*)alloc((size_t)RN * 48 * 4);
  f16* dtin16   = (f16*)alloc((size_t)RN * 32 * 2);
  float* dtraw  = (float*)alloc((size_t)RN * 512 * 4);
  f16* y16      = (f16*)alloc((size_t)RN * 512 * 2);
  f16* cat16    = (f16*)alloc((size_t)RN * 512 * 2);
  float* xfus   = (float*)alloc((size_t)RN * 256 * 4);
  f16* xfus16   = (f16*)alloc((size_t)RN * 256 * 2);
  float* a1     = (float*)alloc((size_t)RN * 256 * 4);
  f16* projWt   = (f16*)alloc(256 * 32 * 2);
  f16* WinT     = (f16*)alloc((size_t)4 * 1024 * 256 * 2);
  f16* WxT      = (f16*)alloc((size_t)4 * 48 * 512 * 2);
  f16* WdtT     = (f16*)alloc((size_t)4 * 512 * 32 * 2);
  f16* WoutT    = (f16*)alloc((size_t)4 * 256 * 512 * 2);
  f16* fusWt    = (f16*)alloc((size_t)256 * 512 * 2);
  f16* agg1t    = (f16*)alloc((size_t)256 * 256 * 2);

  auto blk = [](int n) { return dim3((unsigned)((n + 255) / 256)); };
  auto gemm = [&](const f16* A, int lda, const f16* Bt, int ldb, float* C, int ldc,
                  f16* C16, int ldc16, int M, int N, int K, const float* bias,
                  const float* resid, int ldres, int flags) {
    int tiles = ((M + 31) / 32) * ((N + 15) / 16);
    k_gemm<<<dim3((unsigned)((tiles + 7) / 8)), 256, 0, stream>>>(
        A, lda, Bt, ldb, C, ldc, C16, ldc16, M, N, K, bias, resid, ldres, flags);
  };

  // ---- weight transposes (f32 -> f16 [N][Kpad])
  k_wt<<<blk(256 * 32), 256, 0, stream>>>(proj_W, projWt, 25, 256, 32);
  for (int g = 0; g < 2; ++g)
    for (int l = 0; l < 2; ++l) {
      int wi = g * 2 + l;
      k_wt<<<blk(1024 * 256), 256, 0, stream>>>(grp[g][2] + (size_t)l * 256 * 1024,
                                                WinT + (size_t)wi * 1024 * 256, 256, 1024, 256);
      k_wt<<<blk(48 * 512), 256, 0, stream>>>(grp[g][6] + (size_t)l * 512 * 48,
                                              WxT + (size_t)wi * 48 * 512, 512, 48, 512);
      k_wt<<<blk(512 * 32), 256, 0, stream>>>(grp[g][7] + (size_t)l * 16 * 512,
                                              WdtT + (size_t)wi * 512 * 32, 16, 512, 32);
      k_wt<<<blk(256 * 512), 256, 0, stream>>>(grp[g][11] + (size_t)l * 512 * 256,
                                               WoutT + (size_t)wi * 256 * 512, 512, 256, 512);
    }
  k_wt<<<blk(256 * 512), 256, 0, stream>>>(fus_W, fusWt, 512, 256, 512);
  k_wt<<<blk(256 * 256), 256, 0, stream>>>(agg_W1, agg1t, 256, 256, 256);

  // ---- normalize + pack + cross correlation
  k_norms<<<dim3(QN * PN + SN * PN), 64, 0, stream>>>(query, support, qnorm, snorm);
  k_pack_q<<<blk(PN * QN * CN), 256, 0, stream>>>(query, qnorm, qn16);
  k_pack_su<<<blk(SN * PN * KK2 * CN), 256, 0, stream>>>(support, snorm, su16);
  k_ccorr<<<dim3(1, SN * PN), 256, 0, stream>>>(qn16, su16, cc16);

  // ---- projection + LN/relu/pos -> x0
  gemm(cc16, 32, projWt, 32, a1, 256, nullptr, 0, RN, 256, 32, proj_b, nullptr, 0, 1 | 16);
  k_ln<<<RN, 256, 0, stream>>>(a1, proj_g, proj_bb, spos, x0, nullptr, 1);

  // ---- mamba layer
  auto mamba = [&](const float* xin, float* xout, int g, int layer) {
    const float* ln_g   = grp[g][0] + layer * HIDN;
    const float* ln_b   = grp[g][1] + layer * HIDN;
    const float* b_in   = grp[g][3] + layer * 1024;
    const float* conv_w = grp[g][4] + (size_t)layer * DIN * 4;
    const float* conv_b = grp[g][5] + layer * DIN;
    const float* b_dt   = grp[g][8] + layer * DIN;
    const float* A_log  = grp[g][9] + (size_t)layer * DIN * DSTN;
    const float* Dp     = grp[g][10] + layer * DIN;
    const float* b_out  = grp[g][12] + layer * HIDN;
    int wi = g * 2 + layer;
    const f16* winT  = WinT + (size_t)wi * 1024 * 256;
    const f16* wxT   = WxT + (size_t)wi * 48 * 512;
    const f16* wdtT  = WdtT + (size_t)wi * 512 * 32;
    const f16* woutT = WoutT + (size_t)wi * 256 * 512;

    k_ln<<<RN, 256, 0, stream>>>(xin, ln_g, ln_b, nullptr, nullptr, h16, 0);
    gemm(h16, 256, winT, 256, xz, 1024, nullptr, 0, RN, 1024, 256, b_in, nullptr, 0, 1 | 16);
    k_conv<<<blk(RN * DIN), 256, 0, stream>>>(xz, conv_w, conv_b, xcf, xc16);
    gemm(xc16, 512, wxT, 512, xdbl, 48, nullptr, 0, RN, 48, 512, nullptr, nullptr, 0, 16);
    k_packdt<<<blk(RN * 32), 256, 0, stream>>>(xdbl, dtin16);
    gemm(dtin16, 32, wdtT, 32, dtraw, 512, nullptr, 0, RN, 512, 32, b_dt, nullptr, 0, 1 | 16);
    k_scan<<<BNCH, DIN, 0, stream>>>(xdbl, dtraw, xcf, xz, A_log, Dp, y16);
    gemm(y16, 512, woutT, 512, xout, 256, nullptr, 0, RN, 256, 512, b_out, xin, 256, 1 | 4 | 16);
  };

  // forward stream: x0 -> tA -> xfF
  mamba(x0, tA, 0, 0);
  mamba(tA, xfF, 0, 1);
  // reverse stream: reverse(x0) -> tB -> tA -> tB
  k_rev<<<blk(RN * HIDN), 256, 0, stream>>>(x0, tB);
  mamba(tB, tA, 1, 0);
  mamba(tA, tB, 1, 1);

  // ---- fusion, aggregation, pooling, final LN
  k_cat<<<blk(RN * 512), 256, 0, stream>>>(xfF, tB, cat16);
  gemm(cat16, 512, fusWt, 512, xfus, 256, xfus16, 256, RN, 256, 512, fus_b, nullptr, 0, 1 | 8 | 16);
  gemm(xfus16, 256, agg1t, 256, a1, 256, nullptr, 0, RN, 256, 256, agg_b1, nullptr, 0, 1 | 2 | 16);
  k_pool<<<BNCH, 256, 0, stream>>>(a1, xfus, agg_W2, agg_b2, fin_g, fin_b, out);
}
